// RNN_S2S_53910429499884
// MI455X (gfx1250) — matrix operand order repaired and checked by reading, not compiled, not run on hardware
//
#include <hip/hip_runtime.h>

// ---------------------------------------------------------------------------
// Types for CDNA5 WMMA (wave32) + TDM descriptors
// ---------------------------------------------------------------------------
typedef __bf16 bh;
typedef __attribute__((ext_vector_type(16))) __bf16 v16bf;
typedef __attribute__((ext_vector_type(8)))  __bf16 bh8;
typedef __attribute__((ext_vector_type(8)))  float  v8f;
typedef __attribute__((ext_vector_type(4)))  unsigned int v4u;
typedef __attribute__((ext_vector_type(8)))  int v8i;
typedef __attribute__((ext_vector_type(4)))  int v4i;

union V16 { v16bf v; bh8 h[2]; };

__device__ __forceinline__ v8f wmma_bf16(v16bf a, v16bf b, v8f c) {
  // D = A(16x32 bf16) * B(32x16 bf16) + C(16x16 f32)
  return __builtin_amdgcn_wmma_f32_16x16x32_bf16(
      /*neg_a=*/false, a, /*neg_b=*/false, b,
      /*c_mod=*/(short)0, c, /*reuse_a=*/false, /*reuse_b=*/false);
}

__device__ __forceinline__ float sigm(float x) { return 1.0f / (1.0f + __expf(-x)); }

// Model constants
#define BB   16
#define TT   128
#define DE   128
#define HH   256     // encoder hidden
#define H3   768
#define H2   512     // decoder hidden
#define H6   1536
#define H4   1024
#define VV   32000

// ---------------------------------------------------------------------------
// f32 -> bf16 convert (grid-stride)
// ---------------------------------------------------------------------------
__global__ void k_convert(const float* __restrict__ src, bh* __restrict__ dst, int n) {
  for (int i = blockIdx.x * blockDim.x + threadIdx.x; i < n; i += gridDim.x * blockDim.x)
    dst[i] = (bh)src[i];
}

// ---------------------------------------------------------------------------
// Encoder embedding gather: x_fwd[t*B+b][e], x_bwd uses reversed time
// ---------------------------------------------------------------------------
__global__ void k_embed_enc(const int* __restrict__ tok, const float* __restrict__ emb,
                            bh* __restrict__ xf, bh* __restrict__ xb) {
  int row = blockIdx.x;           // t*16 + b
  int t = row >> 4, b = row & 15;
  int e = threadIdx.x;
  int tf = tok[b * TT + t];
  int tb = tok[b * TT + (TT - 1 - t)];
  xf[row * DE + e] = (bh)emb[tf * DE + e];
  xb[row * DE + e] = (bh)emb[tb * DE + e];
}

// ---------------------------------------------------------------------------
// Decoder embedding: BOS shift + style embedding add
// ---------------------------------------------------------------------------
__global__ void k_embed_dec(const int* __restrict__ tok, const int* __restrict__ labels,
                            const float* __restrict__ emb, const float* __restrict__ style,
                            bh* __restrict__ xd) {
  int row = blockIdx.x;           // t*16 + b
  int t = row >> 4, b = row & 15;
  int e = threadIdx.x;
  int tk = (t == 0) ? 1 : tok[b * TT + t];
  int lb = labels[b];
  xd[row * DE + e] = (bh)(emb[tk * DE + e] + style[lb * DE + e]);
}

// ---------------------------------------------------------------------------
// Generic batched bf16 GEMM: C = scale*(A[M,K] * W[N,K]^T) + bias[N]
// Register-blocked: each wave computes a 64(M) x 32(N) macro-tile
// (4 M-tiles x 2 N-tiles -> 8 WMMAs per K-chunk, A reused 2x, B reused 4x).
// block = 256 threads = 8 waves (8 N-groups). grid=(ceil(N/32/8), M/64, batch)
// ---------------------------------------------------------------------------
__global__ void k_gemm(const bh* __restrict__ A, int lda, long sA,
                       const bh* __restrict__ W, int ldw, long sW,
                       float* __restrict__ Cf, bh* __restrict__ Ch, int ldc, long sC,
                       const float* __restrict__ bias,
                       int N, int K, float scale, int leaky) {
  int tid  = threadIdx.x;
  int lane = tid & 31, wv = tid >> 5;
  int l = lane & 15, hi = lane >> 4;
  int ng = blockIdx.x * 8 + wv;            // N-group of 2 tiles
  if (ng * 32 >= N) return;                // whole-wave uniform exit
  int m0 = blockIdx.y * 64;
  int n0 = ng * 32;
  long zb = blockIdx.z;

  const bh* Ar0 = A + zb * sA + (long)(m0 + l) * lda;
  const bh* Wr0 = W + zb * sW + (long)(n0 + l) * ldw;
  const bh* Wr1 = Wr0 + (long)16 * ldw;

  v8f acc[4][2] = {};
  for (int k = 0; k < K; k += 32) {
    __builtin_prefetch(Wr0 + k + 256, 0, 1);   // global_prefetch_b8 on streamed B
    __builtin_prefetch(Wr1 + k + 256, 0, 1);
    V16 b0, b1;
    b0.h[0] = *(const bh8*)(Wr0 + k + hi * 8);
    b0.h[1] = *(const bh8*)(Wr0 + k + 16 + hi * 8);
    b1.h[0] = *(const bh8*)(Wr1 + k + hi * 8);
    b1.h[1] = *(const bh8*)(Wr1 + k + 16 + hi * 8);
#pragma unroll
    for (int mi = 0; mi < 4; ++mi) {
      const bh* Ar = Ar0 + (long)(mi * 16) * lda;
      V16 a;
      a.h[0] = *(const bh8*)(Ar + k + hi * 8);
      a.h[1] = *(const bh8*)(Ar + k + 16 + hi * 8);
      acc[mi][0] = wmma_bf16(a.v, b0.v, acc[mi][0]);
      acc[mi][1] = wmma_bf16(a.v, b1.v, acc[mi][1]);
    }
  }

  float bv0 = bias ? bias[n0 + l]      : 0.f;
  float bv1 = bias ? bias[n0 + 16 + l] : 0.f;
  long cb = zb * sC;
#pragma unroll
  for (int mi = 0; mi < 4; ++mi)
#pragma unroll
    for (int v = 0; v < 8; ++v) {
      int row = m0 + mi * 16 + v + hi * 8;
      float v0 = acc[mi][0][v] * scale + bv0;
      float v1 = acc[mi][1][v] * scale + bv1;
      if (leaky) {
        v0 = v0 > 0.f ? v0 : 0.01f * v0;
        v1 = v1 > 0.f ? v1 : 0.01f * v1;
      }
      long o = cb + (long)row * ldc + n0 + l;
      if (Ch) { Ch[o] = (bh)v0; Ch[o + 16] = (bh)v1; }
      else    { Cf[o] = v0;     Cf[o + 16] = v1; }
    }
}

// ---------------------------------------------------------------------------
// Logits GEMM (M=2048, N=32000, K=512). The 64x512 bf16 A panel (64 KB,
// contiguous in memory) is staged into LDS by BOTH gfx1250 DMA paths on
// disjoint halves:
//   rows  0..31 (32 KB): Tensor Data Mover (TENSOR_LOAD_TO_LDS, TENSORcnt)
//   rows 32..63 (32 KB): GLOBAL_LOAD_ASYNC_TO_LDS_B128 (ASYNCcnt)
// then shared by all 8 waves; each wave computes 4Mx2N register tiles.
// grid = (32000/256=125, 2048/64=32)
// ---------------------------------------------------------------------------
__global__ void __launch_bounds__(256)
k_gemm_logits(const bh* __restrict__ A, const bh* __restrict__ W,
              float* __restrict__ C, const float* __restrict__ bias) {
  __shared__ __align__(16) bh Ablk[64 * H2];   // 64 KB

  int tid = threadIdx.x;
  int wv  = tid >> 5;
  int m0 = blockIdx.y * 64;

  unsigned lbase = (unsigned)(uintptr_t)(&Ablk[0]);
  unsigned long long gbase = (unsigned long long)(uintptr_t)(A + (long)m0 * H2);

  // --- TDM: first 32 KB (4096 x 8-byte elements, 1 row) ---
  if (wv == 0) {
    unsigned ga_lo = (unsigned)(gbase & 0xffffffffu);
    unsigned ga_hi = (unsigned)(gbase >> 32);
    v4u g0 = { 1u,                                   // count=1 (valid user D#)
               lbase,                                // lds_addr
               ga_lo,                                // global_addr[31:0]
               (ga_hi & 0x01ffffffu) | (2u << 30) }; // global_addr[56:32] | type=2
    v8i g1 = { (int)(3u << 16),                      // wg_mask=0, data_size=8B
               (int)(4096u << 16),                   // tensor_dim0[15:0]=4096
               (int)(1u << 16),                      // tensor_dim0 hi=0, tensor_dim1=1
               (int)(4096u << 16),                   // tensor_dim1 hi=0, tile_dim0=4096
               1,                                    // tile_dim1=1, tile_dim2=0
               4096,                                 // tensor_dim0_stride[31:0]
               0, 0 };                               // stride hi / dim1_stride = 0
    v4i g2 = { 0, 0, 0, 0 };                         // <=2D: groups 2/3 zero
    v4i g3 = { 0, 0, 0, 0 };
    v8i g4 = { 0, 0, 0, 0, 0, 0, 0, 0 };             // VADDR4 unused (NULL)
    __builtin_amdgcn_tensor_load_to_lds(g0, g1, g2, g3, g4, 0);
  }

  // --- async-to-LDS: second 32 KB (bytes 32768..65535) ---
#pragma unroll
  for (int it = 0; it < 8; ++it) {
    unsigned o = 32768u + (unsigned)(tid + it * 256) * 16u;   // byte offset
    unsigned ldsa = lbase + o;
    unsigned long long ga = gbase + o;
    asm volatile("global_load_async_to_lds_b128 %0, %1, off"
                 :: "v"(ldsa), "v"(ga) : "memory");
  }
  asm volatile("s_wait_asynccnt 0x0" ::: "memory");
  if (wv == 0) __builtin_amdgcn_s_wait_tensorcnt(0);
  __syncthreads();

  int lane = tid & 31;
  int l = lane & 15, hi = lane >> 4;
  int n0 = (blockIdx.x * 8 + wv) * 32;

  const bh* Wr0 = W + (long)(n0 + l) * H2;
  const bh* Wr1 = Wr0 + (long)16 * H2;

  v8f acc[4][2] = {};
  for (int k = 0; k < H2; k += 32) {
    __builtin_prefetch(Wr0 + k + 256, 0, 1);
    __builtin_prefetch(Wr1 + k + 256, 0, 1);
    V16 b0, b1;
    b0.h[0] = *(const bh8*)(Wr0 + k + hi * 8);
    b0.h[1] = *(const bh8*)(Wr0 + k + 16 + hi * 8);
    b1.h[0] = *(const bh8*)(Wr1 + k + hi * 8);
    b1.h[1] = *(const bh8*)(Wr1 + k + 16 + hi * 8);
#pragma unroll
    for (int mi = 0; mi < 4; ++mi) {
      V16 a;
      a.h[0] = *(const bh8*)&Ablk[(mi * 16 + l) * H2 + k + hi * 8];
      a.h[1] = *(const bh8*)&Ablk[(mi * 16 + l) * H2 + k + 16 + hi * 8];
      acc[mi][0] = wmma_bf16(a.v, b0.v, acc[mi][0]);
      acc[mi][1] = wmma_bf16(a.v, b1.v, acc[mi][1]);
    }
  }

  float bv0 = bias[n0 + l];
  float bv1 = bias[n0 + 16 + l];
#pragma unroll
  for (int mi = 0; mi < 4; ++mi)
#pragma unroll
    for (int v = 0; v < 8; ++v) {
      int row = m0 + mi * 16 + v + hi * 8;
      long o = (long)row * VV + n0 + l;
      C[o]      = acc[mi][0][v] + bv0;
      C[o + 16] = acc[mi][1][v] + bv1;
    }
}

// ---------------------------------------------------------------------------
// Encoder GRU scans (grid.x = 2: 0=fwd, 1=bwd). block = 768 = 24 waves.
// h in LDS (f32 + bf16 mirror). Per step: A fragments (whole h row, 8 K-chunks)
// cached in VGPRs once, then 2 N-tiles of gh via WMMA per wave.
// ---------------------------------------------------------------------------
__global__ void __launch_bounds__(768)
k_enc_scan(const bh* __restrict__ Uf, const bh* __restrict__ Ub,
           const float* __restrict__ bhf, const float* __restrict__ bhb,
           const float* __restrict__ xpf, const float* __restrict__ xpb,
           float* __restrict__ ghf, float* __restrict__ ghb,
           bh* __restrict__ mem, float* __restrict__ hdec0) {
  __shared__ float hf[BB * HH];   // 16 KB
  __shared__ bh    hb[BB * HH];   // 8 KB

  int dir = blockIdx.x;
  const bh*    U   = dir ? Ub  : Uf;
  const float* bhh = dir ? bhb : bhf;
  const float* xp  = dir ? xpb : xpf;
  float*       gh  = dir ? ghb : ghf;

  int tid = threadIdx.x;
  for (int i = tid; i < BB * HH; i += 768) { hf[i] = 0.f; hb[i] = (bh)0.f; }
  __syncthreads();

  int lane = tid & 31, wv = tid >> 5;
  int l = lane & 15, hi = lane >> 4;

  for (int t = 0; t < TT; ++t) {
    // cache full A row (h[l][0:256]) as 8 fragments
    V16 af[8];
#pragma unroll
    for (int kk = 0; kk < 8; ++kk) {
      af[kk].h[0] = *(const bh8*)&hb[l * HH + kk * 32 + hi * 8];
      af[kk].h[1] = *(const bh8*)&hb[l * HH + kk * 32 + 16 + hi * 8];
    }
    // gh[16,768] = h @ U^T
    for (int nt = wv; nt < H3 / 16; nt += 24) {
      const bh* Wr = U + (long)(nt * 16 + l) * HH;
      v8f acc = {0.f, 0.f, 0.f, 0.f, 0.f, 0.f, 0.f, 0.f};
#pragma unroll
      for (int kk = 0; kk < 8; ++kk) {
        V16 b;
        b.h[0] = *(const bh8*)(Wr + kk * 32 + hi * 8);
        b.h[1] = *(const bh8*)(Wr + kk * 32 + 16 + hi * 8);
        acc = wmma_bf16(af[kk].v, b.v, acc);
      }
#pragma unroll
      for (int v = 0; v < 8; ++v)
        gh[(v + hi * 8) * H3 + nt * 16 + l] = acc[v];
    }
    __threadfence_block();
    __syncthreads();

    const float* xpt = xp + (long)t * BB * H3;
    for (int idx = tid; idx < BB * HH; idx += 768) {
      int b = idx >> 8, j = idx & 255;
      float hv = hf[b * HH + j];
      float r = sigm(xpt[b * H3 + j]          + gh[b * H3 + j]          + bhh[j]);
      float z = sigm(xpt[b * H3 + HH + j]     + gh[b * H3 + HH + j]     + bhh[HH + j]);
      float n = tanhf(xpt[b * H3 + 2*HH + j]  + r * (gh[b * H3 + 2*HH + j] + bhh[2*HH + j]));
      float hn = (1.f - z) * n + z * hv;
      hf[b * HH + j] = hn;
      hb[b * HH + j] = (bh)hn;
      int tt = dir ? (TT - 1 - t) : t;
      mem[(long)(b * TT + tt) * (2 * HH) + dir * HH + j] = (bh)hn;
      if (t == TT - 1)
        hdec0[b * (2 * HH) + dir * HH + j] = hn > 0.f ? hn : 0.01f * hn;
    }
    __syncthreads();
  }
}

// ---------------------------------------------------------------------------
// Decoder GRU scan (1 block, 1024 threads = 32 waves). K split in 2 halves so
// A fragments (8 per half) are cached in VGPRs and reused over 3 N-tiles.
// Writes od -> feat[:, 0:512].
// ---------------------------------------------------------------------------
__global__ void __launch_bounds__(1024)
k_dec_scan(const bh* __restrict__ U, const float* __restrict__ bhh,
           const float* __restrict__ xp, float* __restrict__ gh,
           const float* __restrict__ h0, bh* __restrict__ feat) {
  __shared__ float hf[BB * H2];   // 32 KB
  __shared__ bh    hb[BB * H2];   // 16 KB

  int tid = threadIdx.x;
  for (int i = tid; i < BB * H2; i += 1024) { float v = h0[i]; hf[i] = v; hb[i] = (bh)v; }
  __syncthreads();

  int lane = tid & 31, wv = tid >> 5;
  int l = lane & 15, hi = lane >> 4;

  for (int t = 0; t < TT; ++t) {
    v8f acc[3] = {};
#pragma unroll
    for (int kh = 0; kh < 2; ++kh) {
      V16 af[8];
#pragma unroll
      for (int kk = 0; kk < 8; ++kk) {
        af[kk].h[0] = *(const bh8*)&hb[l * H2 + kh * 256 + kk * 32 + hi * 8];
        af[kk].h[1] = *(const bh8*)&hb[l * H2 + kh * 256 + kk * 32 + 16 + hi * 8];
      }
#pragma unroll
      for (int i = 0; i < 3; ++i) {
        int nt = wv + i * 32;
        const bh* Wr = U + (long)(nt * 16 + l) * H2 + kh * 256;
#pragma unroll
        for (int kk = 0; kk < 8; ++kk) {
          V16 b;
          b.h[0] = *(const bh8*)(Wr + kk * 32 + hi * 8);
          b.h[1] = *(const bh8*)(Wr + kk * 32 + 16 + hi * 8);
          acc[i] = wmma_bf16(af[kk].v, b.v, acc[i]);
        }
      }
    }
#pragma unroll
    for (int i = 0; i < 3; ++i) {
      int nt = wv + i * 32;
#pragma unroll
      for (int v = 0; v < 8; ++v)
        gh[(v + hi * 8) * H6 + nt * 16 + l] = acc[i][v];
    }
    __threadfence_block();
    __syncthreads();

    const float* xpt = xp + (long)t * BB * H6;
    for (int idx = tid; idx < BB * H2; idx += 1024) {
      int b = idx >> 9, j = idx & 511;
      float hv = hf[b * H2 + j];
      float r = sigm(xpt[b * H6 + j]         + gh[b * H6 + j]         + bhh[j]);
      float z = sigm(xpt[b * H6 + H2 + j]    + gh[b * H6 + H2 + j]    + bhh[H2 + j]);
      float n = tanhf(xpt[b * H6 + 2*H2 + j] + r * (gh[b * H6 + 2*H2 + j] + bhh[2*H2 + j]));
      float hn = (1.f - z) * n + z * hv;
      hf[b * H2 + j] = hn;
      hb[b * H2 + j] = (bh)hn;
      feat[(long)(b * TT + t) * H4 + j] = (bh)hn;  // O part of feat
    }
    __syncthreads();
  }
}

// ---------------------------------------------------------------------------
// Row softmax over 128 cols; one wave per row; output bf16.
// ---------------------------------------------------------------------------
__global__ void k_softmax(const float* __restrict__ w, bh* __restrict__ a) {
  int tid = threadIdx.x;
  int lane = tid & 31, wv = tid >> 5;
  int row = blockIdx.x * 8 + wv;
  const float* wr = w + (long)row * TT;
  float v[4];
  float m = -1e30f;
#pragma unroll
  for (int i = 0; i < 4; ++i) { v[i] = wr[lane + 32 * i]; m = fmaxf(m, v[i]); }
#pragma unroll
  for (int off = 16; off > 0; off >>= 1) m = fmaxf(m, __shfl_xor(m, off, 32));
  float s = 0.f;
#pragma unroll
  for (int i = 0; i < 4; ++i) { v[i] = __expf(v[i] - m); s += v[i]; }
#pragma unroll
  for (int off = 16; off > 0; off >>= 1) s += __shfl_xor(s, off, 32);
  float inv = 1.f / s;
#pragma unroll
  for (int i = 0; i < 4; ++i) a[(long)row * TT + lane + 32 * i] = (bh)(v[i] * inv);
}

// ---------------------------------------------------------------------------
// Transpose memory per batch: memT[b][d][t] = mem[b][t][d]
// ---------------------------------------------------------------------------
__global__ void k_transpose(const bh* __restrict__ mem, bh* __restrict__ memT) {
  int bd = blockIdx.x;
  int b = bd >> 9, d = bd & 511;
  int t = threadIdx.x;
  memT[(long)(b * H2 + d) * TT + t] = mem[(long)(b * TT + t) * H2 + d];
}

// ---------------------------------------------------------------------------
// Host launcher
// ---------------------------------------------------------------------------
extern "C" void kernel_launch(void* const* d_in, const int* in_sizes, int n_in,
                              void* d_out, int out_size, void* d_ws, size_t ws_size,
                              hipStream_t stream) {
  const int*   inp_enc = (const int*)d_in[0];
  const int*   inp_dec = (const int*)d_in[1];
  const int*   labels  = (const int*)d_in[2];
  const float* emb     = (const float*)d_in[3];
  const float* style   = (const float*)d_in[4];
  const float* eWf = (const float*)d_in[5];
  const float* eUf = (const float*)d_in[6];
  const float* ebf = (const float*)d_in[7];
  const float* ebhf= (const float*)d_in[8];
  const float* eWb = (const float*)d_in[9];
  const float* eUb = (const float*)d_in[10];
  const float* ebb = (const float*)d_in[11];
  const float* ebhb= (const float*)d_in[12];
  const float* dW  = (const float*)d_in[13];
  const float* dU  = (const float*)d_in[14];
  const float* db  = (const float*)d_in[15];
  const float* dbh = (const float*)d_in[16];
  const float* W1  = (const float*)d_in[17];
  const float* b1  = (const float*)d_in[18];
  const float* W2  = (const float*)d_in[19];
  const float* b2  = (const float*)d_in[20];
  float* out = (float*)d_out;

  // ---- workspace carve (256B aligned) ----
  char* base = (char*)d_ws;
  size_t off = 0;
  auto take = [&](size_t bytes) -> void* {
    void* p = base + off;
    off = (off + bytes + 255) & ~(size_t)255;
    return p;
  };
  bh* eWf_h = (bh*)take(H3 * DE * 2);
  bh* eWb_h = (bh*)take(H3 * DE * 2);
  bh* eUf_h = (bh*)take(H3 * HH * 2);
  bh* eUb_h = (bh*)take(H3 * HH * 2);
  bh* dW_h  = (bh*)take((size_t)H6 * DE * 2);
  bh* dU_h  = (bh*)take((size_t)H6 * H2 * 2);
  bh* W1_h  = (bh*)take((size_t)H2 * H4 * 2);
  bh* W2_h  = (bh*)take((size_t)VV * H2 * 2);
  bh* xef   = (bh*)take((size_t)TT * BB * DE * 2);
  bh* xeb   = (bh*)take((size_t)TT * BB * DE * 2);
  bh* xdh   = (bh*)take((size_t)TT * BB * DE * 2);
  bh* mem_h = (bh*)take((size_t)BB * TT * H2 * 2);
  bh* memT_h= (bh*)take((size_t)BB * H2 * TT * 2);
  bh* a_h   = (bh*)take((size_t)BB * TT * TT * 2);
  bh* feat_h= (bh*)take((size_t)BB * TT * H4 * 2);
  bh* hfeat = (bh*)take((size_t)BB * TT * H2 * 2);
  float* xp_f = (float*)take((size_t)TT * BB * H3 * 4);
  float* xp_b = (float*)take((size_t)TT * BB * H3 * 4);
  float* xp_d = (float*)take((size_t)TT * BB * H6 * 4);
  float* ghf  = (float*)take((size_t)BB * H3 * 4);
  float* ghb  = (float*)take((size_t)BB * H3 * 4);
  float* ghd  = (float*)take((size_t)BB * H6 * 4);
  float* hd0  = (float*)take((size_t)BB * H2 * 4);
  float* attw = (float*)take((size_t)BB * TT * TT * 4);
  (void)ws_size; (void)in_sizes; (void)n_in; (void)out_size;

  // ---- 1. weight conversion f32 -> bf16 ----
  k_convert<<<512,  256, 0, stream>>>(eWf, eWf_h, H3 * DE);
  k_convert<<<512,  256, 0, stream>>>(eWb, eWb_h, H3 * DE);
  k_convert<<<512,  256, 0, stream>>>(eUf, eUf_h, H3 * HH);
  k_convert<<<512,  256, 0, stream>>>(eUb, eUb_h, H3 * HH);
  k_convert<<<512,  256, 0, stream>>>(dW,  dW_h,  H6 * DE);
  k_convert<<<1024, 256, 0, stream>>>(dU,  dU_h,  H6 * H2);
  k_convert<<<1024, 256, 0, stream>>>(W1,  W1_h,  H2 * H4);
  k_convert<<<4096, 256, 0, stream>>>(W2,  W2_h,  VV * H2);

  // ---- 2. embedding gathers ----
  k_embed_enc<<<TT * BB, DE, 0, stream>>>(inp_enc, emb, xef, xeb);
  k_embed_dec<<<TT * BB, DE, 0, stream>>>(inp_dec, labels, emb, style, xdh);

  // ---- 3. input projections (xp = x @ W^T + b) ----
  k_gemm<<<dim3(3, 32, 1), 256, 0, stream>>>(
      xef, DE, 0, eWf_h, DE, 0, xp_f, nullptr, H3, 0, ebf, H3, DE, 1.f, 0);
  k_gemm<<<dim3(3, 32, 1), 256, 0, stream>>>(
      xeb, DE, 0, eWb_h, DE, 0, xp_b, nullptr, H3, 0, ebb, H3, DE, 1.f, 0);
  k_gemm<<<dim3(6, 32, 1), 256, 0, stream>>>(
      xdh, DE, 0, dW_h, DE, 0, xp_d, nullptr, H6, 0, db, H6, DE, 1.f, 0);

  // ---- 4. encoder bidirectional GRU scans ----
  k_enc_scan<<<2, 768, 0, stream>>>(eUf_h, eUb_h, ebhf, ebhb, xp_f, xp_b,
                                    ghf, ghb, mem_h, hd0);

  // ---- 5. decoder GRU scan (writes O into feat[:, 0:512]) ----
  k_dec_scan<<<1, 1024, 0, stream>>>(dU_h, dbh, xp_d, ghd, hd0, feat_h);

  // ---- 6. attention scores: w[b] = O[b] @ mem[b]^T * scale ----
  const float scale = 0.04419417382415922f;  // 1/sqrt(512)
  k_gemm<<<dim3(1, 2, BB), 256, 0, stream>>>(
      feat_h, H4, (long)TT * H4, mem_h, H2, (long)TT * H2,
      attw, nullptr, TT, (long)TT * TT, nullptr, TT, H2, scale, 0);

  // ---- 7. softmax -> bf16 attention weights ----
  k_softmax<<<BB * TT / 8, 256, 0, stream>>>(attw, a_h);

  // ---- 8. transpose memory for context GEMM ----
  k_transpose<<<BB * H2, TT, 0, stream>>>(mem_h, memT_h);

  // ---- 9. context: c[b] = a[b] @ memT[b]^T -> feat[:, 512:1024] ----
  k_gemm<<<dim3(2, 2, BB), 256, 0, stream>>>(
      a_h, TT, (long)TT * TT, memT_h, TT, (long)H2 * TT,
      nullptr, feat_h + H2, H4, (long)TT * H4, nullptr, H2, TT, 1.f, 0);

  // ---- 10. hfeat = leaky_relu(feat @ W1^T + b1), bf16 out ----
  k_gemm<<<dim3(2, 32, 1), 256, 0, stream>>>(
      feat_h, H4, 0, W1_h, H4, 0, nullptr, hfeat, H2, 0, b1, H2, H4, 1.f, 1);

  // ---- 11. logits = hfeat @ W2^T + b2 (dominant GEMM, TDM + async staging) ----
  k_gemm_logits<<<dim3(125, 32, 1), 256, 0, stream>>>(hfeat, W2_h, out, b2);
}
